// SinCAA_3753801417397
// MI455X (gfx1250) — compile-verified, matrix operand order checked by reading, not run on hardware
//
#include <hip/hip_runtime.h>
#include <hip/hip_bf16.h>

// ---------------------------------------------------------------------------
// GraphGPS forward for MI455X (gfx1250, wave32, WMMA).
// GEMMs: v_wmma_f32_16x16x32_f16 with pre-converted f16 B-fragment weights.
// A tiles staged to LDS with global_load_async_to_lds_b64 (ASYNCcnt path).
// ---------------------------------------------------------------------------

#define Cc      128
#define NPG     128
#define Bg      256
#define Nn      (Bg * NPG)       // 32768 nodes
#define E_CNT   262144
#define HEADS   4
#define HD      32
#define NB      ((size_t)Nn * Cc)   // floats per N x C buffer

typedef __attribute__((ext_vector_type(16))) _Float16 v16h;
typedef __attribute__((ext_vector_type(8)))  float    v8f;

union U16 { v16h v; uint4 q[2]; };

static __device__ inline v8f wmma_f16(v16h a, v16h b, v8f c) {
  return __builtin_amdgcn_wmma_f32_16x16x32_f16(false, a, false, b, (short)0, c,
                                                false, false);
}

// Async global->LDS copy of 8 bytes per lane (CDNA5 ASYNCcnt path).
static __device__ inline void async_b64(const float* gsrc, float* lds) {
  unsigned ldsoff = (unsigned)(unsigned long long)lds;   // addr[31:0] = LDS offset
  unsigned long long ga = (unsigned long long)gsrc;
  asm volatile("global_load_async_to_lds_b64 %0, %1, off"
               :: "v"(ldsoff), "v"(ga) : "memory");
}
static __device__ inline void async_wait0() {
  asm volatile("s_wait_asynccnt 0" ::: "memory");
}

// k-index pattern for f16 A/B fragments (16x16x32):
// element i of lane l -> K = (i<8 ? i : i+8) + ((l>=16) ? 8 : 0)

// ---------------------------------------------------------------------------
__global__ __launch_bounds__(256) void k_zero(float* p, int n) {
  int i = blockIdx.x * blockDim.x + threadIdx.x;
  if (i < n) p[i] = 0.0f;
}

__global__ __launch_bounds__(256) void k_copy(float* dst, const float* src) {
  size_t i = (size_t)blockIdx.x * blockDim.x + threadIdx.x;
  ((float4*)dst)[i] = ((const float4*)src)[i];
}

__global__ __launch_bounds__(256) void k_add(float* out, const float* a, const float* b) {
  size_t i = (size_t)blockIdx.x * blockDim.x + threadIdx.x;
  float4 x = ((const float4*)a)[i], y = ((const float4*)b)[i];
  float4 r; r.x = x.x + y.x; r.y = x.y + y.y; r.z = x.z + y.z; r.w = x.w + y.w;
  ((float4*)out)[i] = r;
}

// ---------------------------------------------------------------------------
// Convert weight W[K x stride] (f32, row-major) into WMMA B-fragment f16 cache.
// Layout: [ct][kb][lane(32)][elem(16)].  Column = colOff + ct*16 + (lane&15),
// zero-padded where (ct*16 + lane&15) >= Ncols.
__global__ __launch_bounds__(256) void k_cvtw(const float* W, _Float16* out,
                                              int K, int Ncols, int colOff,
                                              int stride) {
  size_t idx = (size_t)blockIdx.x * blockDim.x + threadIdx.x;
  int i = idx & 15;
  int lane = (idx >> 4) & 31;
  size_t t = idx >> 9;
  int Kt = K >> 5;
  int ct = (int)(t / Kt), kb = (int)(t % Kt);
  int nl = ct * 16 + (lane & 15);
  int k = kb * 32 + (i < 8 ? i : i + 8) + ((lane >> 4) << 3);
  out[idx] = (nl < Ncols) ? (_Float16)W[(size_t)k * stride + colOff + nl]
                          : (_Float16)0.0f;
}

// ---------------------------------------------------------------------------
// x[n,c] = ((ni0+ni1+ni2)/3 + nff @ flw + flb) * mask[n]
__global__ __launch_bounds__(256) void k_embed(const int* nif, const float* nff,
                                               const float* nie, const float* flw,
                                               const float* flb, const float* mask,
                                               float* x) {
  int idx = blockIdx.x * blockDim.x + threadIdx.x;
  int n = idx >> 7, c = idx & 127;
  int i0 = nif[n * 3 + 0], i1 = nif[n * 3 + 1], i2 = nif[n * 3 + 2];
  float e = (nie[(0 * 100 + i0) * Cc + c] + nie[(100 + i1) * Cc + c] +
             nie[(200 + i2) * Cc + c]) * (1.0f / 3.0f);
  float f = flb[c];
#pragma unroll
  for (int j = 0; j < 4; j++) f += nff[n * 4 + j] * flw[j * Cc + c];
  x[idx] = (e + f) * mask[n];
}

// ---------------------------------------------------------------------------
// GINE message: acc[dst] += relu(xm[src] + edge_attr)
__global__ __launch_bounds__(256) void k_scatter(const float* xm, const float* xea,
                                                 float* acc, const int* src,
                                                 const int* dst, const int* ea,
                                                 const float* eemb, const float* mask,
                                                 int mode) {
  int idx = blockIdx.x * blockDim.x + threadIdx.x;   // E * 32 threads
  int e = idx >> 5, cq = (idx & 31) << 2;
  int s = src[e], d = dst[e];
  float4 xs = *(const float4*)(xm + (size_t)s * Cc + cq);
  float4 at;
  if (mode == 0) {
    float em = (mask[s] + mask[d] == 2.0f) ? 0.5f : 0.0f;
    int a0 = ea[e * 2 + 0], a1 = ea[e * 2 + 1];
    const float* p0 = eemb + (size_t)a0 * Cc + cq;
    const float* p1 = eemb + (size_t)(100 + a1) * Cc + cq;
    at.x = (p0[0] + p1[0]) * em; at.y = (p0[1] + p1[1]) * em;
    at.z = (p0[2] + p1[2]) * em; at.w = (p0[3] + p1[3]) * em;
  } else {
    const float* ps = xea + (size_t)s * Cc + cq;
    const float* pd = xea + (size_t)d * Cc + cq;
    at.x = ps[0] + pd[0]; at.y = ps[1] + pd[1];
    at.z = ps[2] + pd[2]; at.w = ps[3] + pd[3];
  }
  float* ap = acc + (size_t)d * Cc + cq;
  unsafeAtomicAdd(ap + 0, fmaxf(xs.x + at.x, 0.0f));
  unsafeAtomicAdd(ap + 1, fmaxf(xs.y + at.y, 0.0f));
  unsafeAtomicAdd(ap + 2, fmaxf(xs.z + at.z, 0.0f));
  unsafeAtomicAdd(ap + 3, fmaxf(xs.w + at.w, 0.0f));
}

// ---------------------------------------------------------------------------
// out[rows x Cout] = epi( A[rows x K] @ W[K x Cout] + bias ), W as f16 frags.
// epi: 0 = bias, 1 = bias + PReLU(*slope), 2 = bias + residual
__global__ __launch_bounds__(256) void k_gemm(const float* A, const _Float16* Wf,
                                              const float* bias, float* out,
                                              const float* res, const float* slope,
                                              int K, int Cout, int epi) {
  __shared__ __align__(16) float Af[16 * 32];
  int lane = threadIdx.x & 31, w = threadIdx.x >> 5;
  int rowBase = blockIdx.x * 16;
  int ct = blockIdx.y * 8 + w;
  int m = lane & 15, hiv = lane >> 4, off = hiv * 8;
  int Kt = K >> 5;
  v8f acc = {0.f, 0.f, 0.f, 0.f, 0.f, 0.f, 0.f, 0.f};
  int e = threadIdx.x * 2;          // float index inside 16x32 tile
  int sr = e >> 5, sc = e & 31;
  const _Float16* wp = Wf + ((size_t)ct * Kt * 32 + lane) * 16;

  for (int kb = 0; kb < Kt; kb++) {
    __syncthreads();
    async_b64(A + (size_t)(rowBase + sr) * K + kb * 32 + sc, Af + e);
    async_wait0();
    __syncthreads();
    __builtin_prefetch(wp + (size_t)(kb + 1) * 512, 0, 1);
    U16 ua;
    const float* ap = Af + m * 32 + off;
#pragma unroll
    for (int i = 0; i < 8; i++) ua.v[i] = (_Float16)ap[i];
#pragma unroll
    for (int i = 0; i < 8; i++) ua.v[8 + i] = (_Float16)ap[16 + i];
    U16 ub;
    const uint4* bp = (const uint4*)(wp + (size_t)kb * 512);
    ub.q[0] = bp[0];
    ub.q[1] = bp[1];
    acc = wmma_f16(ua.v, ub.v, acc);
  }
  int ncol = ct * 16 + m;
  if (ncol >= Cout) return;
  float bv = bias[ncol];
  float sl = (epi == 1) ? slope[0] : 0.0f;
#pragma unroll
  for (int r = 0; r < 8; r++) {
    int row = rowBase + r + hiv * 8;
    float v = acc[r] + bv;
    if (epi == 1) v = (v >= 0.0f) ? v : sl * v;
    if (epi == 2) v += res[(size_t)row * Cout + ncol];
    out[(size_t)row * Cout + ncol] = v;
  }
}

// ---------------------------------------------------------------------------
// GraphNorm: out = (v - a*mean) * rsqrt(var + 1e-5) * g + b, v = in1 (+ in2)
__global__ __launch_bounds__(128) void k_graphnorm(const float* in1, const float* in2,
                                                   const float* pa, const float* pb,
                                                   const float* pg, float* out) {
  int b = blockIdx.x, c = threadIdx.x;
  const float* p1 = in1 + (size_t)b * NPG * Cc + c;
  const float* p2 = in2 ? in2 + (size_t)b * NPG * Cc + c : nullptr;
  float s = 0.0f;
  for (int n = 0; n < NPG; n++) {
    float v = p1[n * Cc];
    if (p2) v += p2[n * Cc];
    s += v;
  }
  float mean = s * (1.0f / NPG);
  float a = pa[c];
  float var = 0.0f;
  for (int n = 0; n < NPG; n++) {
    float v = p1[n * Cc];
    if (p2) v += p2[n * Cc];
    v -= a * mean;
    var += v * v;
  }
  float inv = rsqrtf(var * (1.0f / NPG) + 1e-5f);
  float g = pg[c], bb = pb[c];
  float* po = out + (size_t)b * NPG * Cc + c;
  for (int n = 0; n < NPG; n++) {
    float v = p1[n * Cc];
    if (p2) v += p2[n * Cc];
    v -= a * mean;
    po[n * Cc] = v * inv * g + bb;
  }
}

// ---------------------------------------------------------------------------
// MHA per (graph, head): S = Q K^T * scale, softmax, O = P V.  WMMA end to end.
__global__ __launch_bounds__(256) void k_attn(const float* Q, const float* Kt_,
                                              const float* V, float* O) {
  __shared__ __align__(16) _Float16 Kl[NPG * HD];   // [node][d]
  __shared__ __align__(16) _Float16 Vl[HD * NPG];   // [d][node] (transposed)
  __shared__ __align__(16) _Float16 Pl[8 * 16 * NPG];
  int b = blockIdx.x >> 2, h = blockIdx.x & 3;
  int nodeBase = b * NPG;
  int lane = threadIdx.x & 31, w = threadIdx.x >> 5;
  int m = lane & 15, hiv = lane >> 4, off = hiv * 8;
  int rowBase = w * 16;

  // stage K [node][d] and V transposed [d][node] as f16
  int e0 = threadIdx.x * 16;
#pragma unroll
  for (int i = 0; i < 16; i++) {
    int e = e0 + i;
    int nn = e >> 5, d = e & 31;
    Kl[e] = (_Float16)Kt_[(size_t)(nodeBase + nn) * Cc + h * HD + d];
    Vl[d * NPG + nn] = (_Float16)V[(size_t)(nodeBase + nn) * Cc + h * HD + d];
  }
  __syncthreads();

  // Q fragment (A: 16x32, K-dim = head dim = 32 -> one WMMA step)
  v16h qa;
  {
    int node = nodeBase + rowBase + m;
    const float* qp = Q + (size_t)node * Cc + h * HD;
    float4 a0 = *(const float4*)(qp + off);
    float4 a1 = *(const float4*)(qp + off + 4);
    float4 a2 = *(const float4*)(qp + off + 16);
    float4 a3 = *(const float4*)(qp + off + 20);
    qa[0] = (_Float16)a0.x; qa[1] = (_Float16)a0.y;
    qa[2] = (_Float16)a0.z; qa[3] = (_Float16)a0.w;
    qa[4] = (_Float16)a1.x; qa[5] = (_Float16)a1.y;
    qa[6] = (_Float16)a1.z; qa[7] = (_Float16)a1.w;
    qa[8] = (_Float16)a2.x; qa[9] = (_Float16)a2.y;
    qa[10] = (_Float16)a2.z; qa[11] = (_Float16)a2.w;
    qa[12] = (_Float16)a3.x; qa[13] = (_Float16)a3.y;
    qa[14] = (_Float16)a3.z; qa[15] = (_Float16)a3.w;
  }

  v8f st[8];
#pragma unroll
  for (int ct = 0; ct < 8; ct++) {
    U16 ub;
    int j = ct * 16 + m;                  // key node column
    ub.q[0] = *(const uint4*)(Kl + j * HD + off);
    ub.q[1] = *(const uint4*)(Kl + j * HD + off + 16);
    v8f z = {0.f, 0.f, 0.f, 0.f, 0.f, 0.f, 0.f, 0.f};
    st[ct] = wmma_f16(qa, ub.v, z);
  }

  const float scale = 0.17677669529663687f;  // 1/sqrt(32)
#pragma unroll
  for (int r = 0; r < 8; r++) {
    float mx = -1e30f;
#pragma unroll
    for (int ct = 0; ct < 8; ct++) mx = fmaxf(mx, st[ct][r]);
#pragma unroll
    for (int msk = 1; msk < 16; msk <<= 1) mx = fmaxf(mx, __shfl_xor(mx, msk));
    float mxs = mx * scale;
    float pv[8];
    float sum = 0.0f;
#pragma unroll
    for (int ct = 0; ct < 8; ct++) {
      float ev = __expf(st[ct][r] * scale - mxs);
      pv[ct] = ev;
      sum += ev;
    }
#pragma unroll
    for (int msk = 1; msk < 16; msk <<= 1) sum += __shfl_xor(sum, msk);
    float inv = 1.0f / sum;
    int row = r + hiv * 8;
#pragma unroll
    for (int ct = 0; ct < 8; ct++)
      Pl[(w * 16 + row) * NPG + ct * 16 + m] = (_Float16)(pv[ct] * inv);
  }
  __syncthreads();

  // O = P @ V  (K-dim 128, two 16-wide output col tiles)
#pragma unroll
  for (int jt = 0; jt < 2; jt++) {
    v8f oacc = {0.f, 0.f, 0.f, 0.f, 0.f, 0.f, 0.f, 0.f};
    int j = jt * 16 + m;
    for (int kb = 0; kb < NPG; kb += 32) {
      U16 ua, ub;
      ua.q[0] = *(const uint4*)(Pl + (w * 16 + m) * NPG + kb + off);
      ua.q[1] = *(const uint4*)(Pl + (w * 16 + m) * NPG + kb + off + 16);
      ub.q[0] = *(const uint4*)(Vl + j * NPG + kb + off);
      ub.q[1] = *(const uint4*)(Vl + j * NPG + kb + off + 16);
      oacc = wmma_f16(ua.v, ub.v, oacc);
    }
#pragma unroll
    for (int r = 0; r < 8; r++) {
      int row = rowBase + r + hiv * 8;
      O[(size_t)(nodeBase + row) * Cc + h * HD + j] = oacc[r];
    }
  }
}

// ---------------------------------------------------------------------------
// Reconstruction head + masked CE + accuracy, fused.  Wf = per-segment f16
// fragments (2 segs x 7 col tiles x 4 k tiles), columns >=100 zero-padded.
__global__ __launch_bounds__(256) void k_recloss(const float* X, const int* src,
                                                 const int* dst, const int* labels,
                                                 int labStride, const float* mask,
                                                 const _Float16* Wf, const float* bias,
                                                 float* accum, int mode) {
  __shared__ __align__(16) _Float16 Al[8 * 16 * Cc];
  int lane = threadIdx.x & 31, w = threadIdx.x >> 5;
  int rowBase = blockIdx.x * 128 + w * 16;
  int m = lane & 15, hiv = lane >> 4, off = hiv * 8;

  // stage this wave's 16x128 feature strip as f16 (float4 loads, b64 stores)
  {
    int row = rowBase + m;
    int k0 = hiv * 64;
    const float* xs;
    const float* xd = nullptr;
    if (mode == 0) {
      xs = X + (size_t)row * Cc;
    } else {
      xs = X + (size_t)src[row] * Cc;
      xd = X + (size_t)dst[row] * Cc;
    }
    _Float16* dp = Al + (w * 16 + m) * Cc + k0;
#pragma unroll
    for (int t = 0; t < 16; t++) {
      float4 v = *(const float4*)(xs + k0 + t * 4);
      if (mode) {
        float4 u = *(const float4*)(xd + k0 + t * 4);
        v.x += u.x; v.y += u.y; v.z += u.z; v.w += u.w;
      }
      union { _Float16 h[4]; uint2 u2; } pk;
      pk.h[0] = (_Float16)v.x; pk.h[1] = (_Float16)v.y;
      pk.h[2] = (_Float16)v.z; pk.h[3] = (_Float16)v.w;
      *(uint2*)(dp + t * 4) = pk.u2;
    }
  }
  __syncthreads();

  float rmask[8];
#pragma unroll
  for (int r = 0; r < 8; r++) {
    int row = rowBase + r + hiv * 8;
    float rm;
    if (mode == 0) rm = (mask[row] < 1.0f) ? 1.0f : 0.0f;
    else {
      int s_ = src[row], d_ = dst[row];
      rm = ((mask[s_] + mask[d_]) == 2.0f) ? 0.0f : 1.0f;
    }
    rmask[r] = rm;
  }

  float lce = 0.0f, lcorr = 0.0f, lcnt = 0.0f;
  for (int seg = 0; seg < 2; seg++) {
    const _Float16* Ws = Wf + (size_t)seg * 7 * 4 * 512;
    float sm[8], ss[8], labv[8], bestv[8];
    int bestc[8], lab[8];
#pragma unroll
    for (int r = 0; r < 8; r++) {
      sm[r] = -1e30f; ss[r] = 0.0f; labv[r] = -1e30f;
      bestv[r] = -1e30f; bestc[r] = 1 << 20;
      lab[r] = labels[(size_t)(rowBase + r + hiv * 8) * labStride + seg];
    }
    for (int ct = 0; ct < 7; ct++) {
      int nl = ct * 16 + m;
      bool valid = (nl < 100);
      int ncol = seg * 100 + nl;
      v8f acc = {0.f, 0.f, 0.f, 0.f, 0.f, 0.f, 0.f, 0.f};
      for (int kb = 0; kb < 4; kb++) {
        U16 ua, ub;
        ua.q[0] = *(const uint4*)(Al + (w * 16 + m) * Cc + kb * 32 + off);
        ua.q[1] = *(const uint4*)(Al + (w * 16 + m) * Cc + kb * 32 + off + 16);
        const uint4* bp = (const uint4*)(Ws + (((size_t)ct * 4 + kb) * 32 + lane) * 16);
        ub.q[0] = bp[0];
        ub.q[1] = bp[1];
        acc = wmma_f16(ua.v, ub.v, acc);
      }
      float bv = valid ? bias[ncol] : 0.0f;
#pragma unroll
      for (int r = 0; r < 8; r++) {
        float v = valid ? (acc[r] + bv) : -1e30f;
        float nm = fmaxf(sm[r], v);
        ss[r] = ss[r] * __expf(sm[r] - nm) + __expf(v - nm);
        sm[r] = nm;
        if (valid && nl == lab[r]) labv[r] = v;
        if (v > bestv[r]) { bestv[r] = v; bestc[r] = nl; }
      }
    }
#pragma unroll
    for (int r = 0; r < 8; r++) {
      float M = sm[r], S = ss[r], L = labv[r], Bv = bestv[r];
      int Bc = bestc[r];
#pragma unroll
      for (int msk = 1; msk < 16; msk <<= 1) {
        float oM = __shfl_xor(M, msk);
        float oS = __shfl_xor(S, msk);
        float oL = __shfl_xor(L, msk);
        float oBv = __shfl_xor(Bv, msk);
        int oBc = __shfl_xor(Bc, msk);
        float nm = fmaxf(M, oM);
        S = S * __expf(M - nm) + oS * __expf(oM - nm);
        M = nm;
        L = fmaxf(L, oL);
        if (oBv > Bv || (oBv == Bv && oBc < Bc)) { Bv = oBv; Bc = oBc; }
      }
      if (m == 0) {
        float ce = M + __logf(S) - L;
        float rm = rmask[r];
        lce += ce * rm;
        lcorr += ((Bc == lab[r]) ? 1.0f : 0.0f) * rm;
        if (seg == 0) lcnt += rm;
      }
    }
  }
  if (m == 0) {
    unsafeAtomicAdd(&accum[0], lce);
    unsafeAtomicAdd(&accum[1], lcorr);
    unsafeAtomicAdd(&accum[2], lcnt);
  }
}

// ---------------------------------------------------------------------------
__global__ __launch_bounds__(128) void k_retemb(const float* x, float* out) {
  int b = blockIdx.x, c = threadIdx.x;
  float s = 0.0f;
  for (int n = 0; n < NPG; n++) s += x[(size_t)(b * NPG + n) * Cc + c];
  out[b * Cc + c] = s;
}

__global__ __launch_bounds__(256) void k_sim(const float* aaE, const float* nbrE,
                                             const float* w, const float* bsc,
                                             float* out) {
  int b = threadIdx.x;
  float s = bsc[0];
  for (int c = 0; c < Cc; c++)
    s += aaE[b * Cc + c] * w[c] + nbrE[b * Cc + c] * w[Cc + c];
  out[b] = 1.0f / (1.0f + __expf(-s));
}

__global__ void k_final(const float* acc, float* loss, float* accuracy) {
  float L = 0.0f, A = 0.0f;
  for (int g = 0; g < 3; g++) {
    const float* a = acc + g * 6;
    float nden = fmaxf(2.0f * a[2], 1.0f);
    float eden = fmaxf(2.0f * a[5], 1.0f);
    L += a[0] / nden + a[3] / eden;
    A += (a[1] / nden + a[4] / eden) * 0.5f;
  }
  *loss = L;
  *accuracy = A * (1.0f / 3.0f);
}

// ---------------------------------------------------------------------------
struct LayerP {
  const float *wkb, *wob, *wqb, *wvb;                  // biases (f32)
  const _Float16 *wkf, *wof, *wqf, *wvf;               // weight fragments (f16)
  const float *g0a1, *g0l1b, *g0l2b, *g1a1, *g1l1b, *g1l2b;
  const _Float16 *g0l1f, *g0l2f, *g1l1f, *g1l2f;
  const float *mlpa, *m1b, *m2b;
  const _Float16 *m1f, *m2f;
  const float *n1a, *n1b, *n1g, *n2a, *n2b, *n2g, *n3a, *n3b, *n3g;
};

extern "C" void kernel_launch(void* const* d_in, const int* in_sizes, int n_in,
                              void* d_out, int out_size, void* d_ws, size_t ws_size,
                              hipStream_t stream) {
  (void)in_sizes; (void)n_in; (void)out_size; (void)ws_size;
  auto F = [&](int i) { return (const float*)d_in[i]; };
  auto I = [&](int i) { return (const int*)d_in[i]; };

  // params flattened with dict keys sorted (jax pytree order)
  const float* edge_emb = F(0);
  const float* erec_b = F(1); const float* erec_w = F(2);
  const float* fl_b = F(3);   const float* fl_w = F(4);
  const float* nie = F(101);
  const float* rec_b = F(102); const float* rec_w = F(103);
  const float* sim_b = F(104); const float* sim_w = F(105);

  float* ws = (float*)d_ws;
  auto WB = [&](int i) { return ws + (size_t)i * NB; };
  float* bufH = ws + (size_t)7 * NB;                 // N x 256 MLP hidden
  float* accum = bufH + (size_t)Nn * 256;            // loss accumulators
  _Float16* wf = (_Float16*)(accum + 32);            // f16 weight fragments
  float* out = (float*)d_out;

  // ---- weight fragment conversion (once; shared by all 3 graphs) ----------
  size_t wo = 0;
  auto alloc = [&](int Kt, int Nt) { _Float16* p = wf + wo;
                                     wo += (size_t)Kt * Nt * 512; return p; };
  auto cv = [&](const float* W, _Float16* dst, int K, int Ncols, int colOff,
                int stride, int Nt) {
    int Kt = K >> 5;
    int blocks = (Kt * Nt * 512) / 256;
    k_cvtw<<<blocks, 256, 0, stream>>>(W, dst, K, Ncols, colOff, stride);
  };

  LayerP LP[3];
  for (int L = 0; L < 3; L++) {
    int o = 5 + 32 * L;
    LayerP& p = LP[L];
    p.wkb = F(o + 0); p.wob = F(o + 2); p.wqb = F(o + 4); p.wvb = F(o + 6);
    p.g0a1 = F(o + 8); p.g0l1b = F(o + 9); p.g0l2b = F(o + 11);
    p.g1a1 = F(o + 13); p.g1l1b = F(o + 14); p.g1l2b = F(o + 16);
    p.mlpa = F(o + 18); p.m1b = F(o + 19); p.m2b = F(o + 21);
    p.n1a = F(o + 23); p.n1b = F(o + 24); p.n1g = F(o + 25);
    p.n2a = F(o + 26); p.n2b = F(o + 27); p.n2g = F(o + 28);
    p.n3a = F(o + 29); p.n3b = F(o + 30); p.n3g = F(o + 31);

    _Float16* t;
    t = alloc(4, 8);  cv(F(o + 1), t, 128, 128, 0, 128, 8);  p.wkf = t;
    t = alloc(4, 8);  cv(F(o + 3), t, 128, 128, 0, 128, 8);  p.wof = t;
    t = alloc(4, 8);  cv(F(o + 5), t, 128, 128, 0, 128, 8);  p.wqf = t;
    t = alloc(4, 8);  cv(F(o + 7), t, 128, 128, 0, 128, 8);  p.wvf = t;
    t = alloc(4, 8);  cv(F(o + 10), t, 128, 128, 0, 128, 8); p.g0l1f = t;
    t = alloc(4, 8);  cv(F(o + 12), t, 128, 128, 0, 128, 8); p.g0l2f = t;
    t = alloc(4, 8);  cv(F(o + 15), t, 128, 128, 0, 128, 8); p.g1l1f = t;
    t = alloc(4, 8);  cv(F(o + 17), t, 128, 128, 0, 128, 8); p.g1l2f = t;
    t = alloc(4, 16); cv(F(o + 20), t, 128, 256, 0, 256, 16); p.m1f = t;
    t = alloc(8, 8);  cv(F(o + 22), t, 256, 128, 0, 128, 8); p.m2f = t;
  }
  _Float16* recf = alloc(4, 14);   // 2 segments x 7 tiles
  cv(rec_w, recf, 128, 100, 0, 200, 7);
  cv(rec_w, recf + (size_t)7 * 4 * 512, 128, 100, 100, 200, 7);
  _Float16* erecf = alloc(4, 14);
  cv(erec_w, erecf, 128, 100, 0, 200, 7);
  cv(erec_w, erecf + (size_t)7 * 4 * 512, 128, 100, 100, 200, 7);

  k_zero<<<1, 32, 0, stream>>>(accum, 24);

  dim3 gg(Nn / 16, 1), gg2(Nn / 16, 2);
  const int CPB = NB / 1024;

  for (int g = 0; g < 3; g++) {
    int gb = 106 + 5 * g;
    const int* nif = I(gb + 0);
    const float* nff = F(gb + 1);
    const int* ea = I(gb + 2);
    const int* ei = I(gb + 3);
    const float* mask = F(gb + 4);
    const int* src = ei;
    const int* dst = ei + E_CNT;

    float* x = WB(0);
    k_embed<<<(Nn * Cc) / 256, 256, 0, stream>>>(nif, nff, nie, fl_w, fl_b, mask, x);

    for (int L = 0; L < 3; L++) {
      const LayerP& p = LP[L];
      int mode = (L == 0) ? 0 : 1;
      // GINE conv 1 (input x)
      k_copy<<<CPB, 256, 0, stream>>>(WB(1), x);
      k_scatter<<<(E_CNT * 32) / 256, 256, 0, stream>>>(x, x, WB(1), src, dst, ea,
                                                        edge_emb, mask, mode);
      k_gemm<<<gg, 256, 0, stream>>>(WB(1), p.g0l1f, p.g0l1b, WB(2), nullptr,
                                     p.g0a1, 128, 128, 1);
      k_gemm<<<gg, 256, 0, stream>>>(WB(2), p.g0l2f, p.g0l2b, WB(3), nullptr,
                                     nullptr, 128, 128, 0);
      // GINE conv 2 (input WB(3), edge attr from x)
      k_copy<<<CPB, 256, 0, stream>>>(WB(1), WB(3));
      k_scatter<<<(E_CNT * 32) / 256, 256, 0, stream>>>(WB(3), x, WB(1), src, dst,
                                                        ea, edge_emb, mask, mode);
      k_gemm<<<gg, 256, 0, stream>>>(WB(1), p.g1l1f, p.g1l1b, WB(2), nullptr,
                                     p.g1a1, 128, 128, 1);
      k_gemm<<<gg, 256, 0, stream>>>(WB(2), p.g1l2f, p.g1l2b, WB(1), nullptr,
                                     nullptr, 128, 128, 0);
      // hn = norm1(h + x)
      k_graphnorm<<<Bg, 128, 0, stream>>>(WB(1), x, p.n1a, p.n1b, p.n1g, WB(3));
      // attention branch
      k_gemm<<<gg, 256, 0, stream>>>(x, p.wqf, p.wqb, WB(4), nullptr, nullptr, 128, 128, 0);
      k_gemm<<<gg, 256, 0, stream>>>(x, p.wkf, p.wkb, WB(5), nullptr, nullptr, 128, 128, 0);
      k_gemm<<<gg, 256, 0, stream>>>(x, p.wvf, p.wvb, WB(6), nullptr, nullptr, 128, 128, 0);
      k_attn<<<Bg * HEADS, 256, 0, stream>>>(WB(4), WB(5), WB(6), WB(1));
      k_gemm<<<gg, 256, 0, stream>>>(WB(1), p.wof, p.wob, WB(2), nullptr, nullptr, 128, 128, 0);
      k_graphnorm<<<Bg, 128, 0, stream>>>(WB(2), x, p.n2a, p.n2b, p.n2g, WB(4));
      // out0 = hn + ha ; MLP ; norm3
      k_add<<<CPB, 256, 0, stream>>>(WB(5), WB(3), WB(4));
      k_gemm<<<gg2, 256, 0, stream>>>(WB(5), p.m1f, p.m1b, bufH, nullptr, p.mlpa,
                                      128, 256, 1);
      k_gemm<<<gg, 256, 0, stream>>>(bufH, p.m2f, p.m2b, WB(6), WB(5), nullptr,
                                     256, 128, 2);
      k_graphnorm<<<Bg, 128, 0, stream>>>(WB(6), nullptr, p.n3a, p.n3b, p.n3g, x);
    }

    if (g == 0) k_retemb<<<Bg, 128, 0, stream>>>(x, out);
    if (g == 1) k_retemb<<<Bg, 128, 0, stream>>>(x, out + Bg * Cc);
    k_recloss<<<Nn / 128, 256, 0, stream>>>(x, nullptr, nullptr, nif, 3, mask,
                                            recf, rec_b, accum + g * 6, 0);
    k_recloss<<<E_CNT / 128, 256, 0, stream>>>(x, src, dst, ea, 2, mask,
                                               erecf, erec_b, accum + g * 6 + 3, 1);
  }

  k_sim<<<1, 256, 0, stream>>>(out, out + Bg * Cc, sim_w, sim_b, out + 65537);
  k_final<<<1, 1, 0, stream>>>(accum, out + 65536, out + 65793);
}
